// FRGinPredictorAgent_34256659153346
// MI455X (gfx1250) — matrix operand order repaired and checked by reading, not compiled, not run on hardware
//
#include <hip/hip_runtime.h>
#include <hip/hip_bf16.h>
#include <math.h>

typedef _Float16 f16;
typedef __attribute__((ext_vector_type(16))) _Float16 v16h;
typedef __attribute__((ext_vector_type(8)))  float    v8f;
typedef __attribute__((ext_vector_type(4)))  float    v4f;

#define N_NODES  100000
#define N_EDGES  1600000
#define N_GRAPHS 256
#define IN_DIM   6
#define HID      32
#define DEC      16

// ---------------------------------------------------------------------------
// prep: fold BN into scale/shift; convert weights to f16 pre-swizzled into the
// exact WMMA B-operand lane layout:
//   entry [t*512 + L*16 + h]  =  W[k][n],  k = (L&16 ? 8:0)+(h<8 ? h : h+8),  n = (L&15)+16t
// so each wave lane L loads its whole B fragment as one contiguous v16h.
// ---------------------------------------------------------------------------
struct PrepArgs {
    const float* wa[3];
    const float* wb[3];
    const float* bb[3];
    const float* g[3];
    const float* bt[3];
    const float* m[3];
    const float* v[3];
    int din[3];
};

__global__ __launch_bounds__(256) void prep_kernel(PrepArgs p, f16* waS, f16* wbS,
                                                   float* scale, float* shift) {
    int l = blockIdx.x;
    for (int idx = threadIdx.x; idx < HID * HID; idx += blockDim.x) {
        int t = idx >> 9;          // column tile 0/1
        int L = (idx >> 4) & 31;   // lane
        int h = idx & 15;          // half index within fragment
        int k = ((L & 16) ? 8 : 0) + ((h < 8) ? h : h + 8);
        int n = (L & 15) + t * 16;
        waS[l * HID * HID + idx] = (k < p.din[l]) ? (f16)p.wa[l][k * HID + n] : (f16)0.0f;
        wbS[l * HID * HID + idx] = (f16)p.wb[l][k * HID + n];
    }
    if (threadIdx.x < HID) {
        int n = threadIdx.x;
        float inv = rsqrtf(p.v[l][n] + 1e-5f);
        float sc  = p.g[l][n] * inv;
        scale[l * HID + n] = sc;
        // out = (h_lin + bb - m)*sc + bt  ->  h_lin*sc + shift
        shift[l * HID + n] = (p.bb[l][n] - p.m[l][n]) * sc + p.bt[l][n];
    }
}

// ---------------------------------------------------------------------------
// pad x (N,6) -> h0 (N,32) with zero padding
// ---------------------------------------------------------------------------
__global__ __launch_bounds__(256) void pad_x_kernel(const float* __restrict__ x,
                                                    float* __restrict__ h0) {
    int wave = (blockIdx.x * blockDim.x + threadIdx.x) >> 5;
    int lane = threadIdx.x & 31;
    if (wave >= N_NODES) return;
    h0[(long)wave * HID + lane] = (lane < IN_DIM) ? x[(long)wave * IN_DIM + lane] : 0.0f;
}

// ---------------------------------------------------------------------------
// generic float zero / copy
// ---------------------------------------------------------------------------
__global__ __launch_bounds__(256) void zero_kernel(float* __restrict__ p, long n) {
    long i = (long)blockIdx.x * blockDim.x + threadIdx.x;
    if (i < n) p[i] = 0.0f;
}

__global__ __launch_bounds__(256) void copy_kernel(const float* __restrict__ s,
                                                   float* __restrict__ d, long n) {
    long i = (long)blockIdx.x * blockDim.x + threadIdx.x;
    if (i < n) d[i] = s[i];
}

// ---------------------------------------------------------------------------
// edge scatter: agg[dst,:] += h[src,:]  (wave per edge, lane = channel)
// agg is pre-initialized with h itself, so after this pass agg = h + sum(nbrs)
// ---------------------------------------------------------------------------
__global__ __launch_bounds__(256) void edge_scatter_kernel(const int* __restrict__ ei,
                                                           const float* __restrict__ h,
                                                           float* __restrict__ agg) {
    int wave = (blockIdx.x * blockDim.x + threadIdx.x) >> 5;
    int lane = threadIdx.x & 31;
    if (wave >= N_EDGES) return;
    int src = ei[wave];
    int dst = ei[N_EDGES + wave];
    atomicAdd(&agg[(long)dst * HID + lane], h[(long)src * HID + lane]);
}

// ---------------------------------------------------------------------------
// GIN MLP: hout = BN(ReLU(agg@Wa + ba)@Wb + bb)   (agg already = h + sum nbrs)
// one wave32 per 16 nodes; 4x v_wmma_f32_16x16x32_f16 per wave
// ---------------------------------------------------------------------------
__global__ __launch_bounds__(256) void gin_mlp_kernel(const float* __restrict__ agg,
                                                      const f16* __restrict__ waS,
                                                      const f16* __restrict__ wbS,
                                                      const float* __restrict__ ba,
                                                      const float* __restrict__ scale,
                                                      const float* __restrict__ shift,
                                                      float* __restrict__ hout) {
    __shared__ float tile[8][16][36];   // per-wave 16x32 restage; 36-stride keeps
                                        // rows 16B-aligned, 36r mod 64 permutes banks

    const int waveId   = threadIdx.x >> 5;
    const int lane     = threadIdx.x & 31;
    const int row_base = (blockIdx.x * 8 + waveId) * 16;
    const int r        = lane & 15;            // matrix row (A) / column (B,D)
    const int kbaseA   = (lane < 16) ? 0 : 8;  // A-layout K base for this half-wave
    const int rofsD    = (lane < 16) ? 0 : 8;  // D-layout row base

    // ---- load A = agg tile (f32, vectorized) -> f16, ISA A-layout ----
    // lane's 16 K-values = two contiguous runs of 8: [kb..kb+7], [kb+16..kb+23]
    const int  rowA   = row_base + r;
    const bool validA = rowA < N_NODES;
    const v4f  zf     = {0.0f, 0.0f, 0.0f, 0.0f};
    const v4f* g4     = (const v4f*)(agg + (long)rowA * HID);
    const int  q0     = kbaseA >> 2;           // 0 or 2
    v4f p0 = validA ? g4[q0]     : zf;
    v4f p1 = validA ? g4[q0 + 1] : zf;
    v4f p2 = validA ? g4[q0 + 4] : zf;         // +16 floats
    v4f p3 = validA ? g4[q0 + 5] : zf;
    v16h a;
    #pragma unroll
    for (int i = 0; i < 4; ++i) {
        a[i]      = (f16)p0[i];
        a[i + 4]  = (f16)p1[i];
        a[i + 8]  = (f16)p2[i];
        a[i + 12] = (f16)p3[i];
    }

    // ---- B fragments: pre-swizzled, one contiguous v16h per lane per tile ----
    const v16h* waV = (const v16h*)waS;
    const v16h* wbV = (const v16h*)wbS;
    v16h b0 = waV[lane];
    v16h b1 = waV[32 + lane];

    v8f c0 = {}, c1 = {};
    c0 = __builtin_amdgcn_wmma_f32_16x16x32_f16(false, a, false, b0, (short)0, c0, false, false);
    c1 = __builtin_amdgcn_wmma_f32_16x16x32_f16(false, a, false, b1, (short)0, c1, false, false);

    // ---- bias + ReLU in D layout, restage to LDS as a plain 16x32 tile ----
    const float bias0 = ba[r];
    const float bias1 = ba[r + 16];
    #pragma unroll
    for (int i = 0; i < 8; ++i) {
        tile[waveId][rofsD + i][r]      = fmaxf(c0[i] + bias0, 0.0f);
        tile[waveId][rofsD + i][r + 16] = fmaxf(c1[i] + bias1, 0.0f);
    }
    // same-wave LDS ops are ordered; compiler inserts s_wait_dscnt for the reads.

    // ---- reload as A-layout f16 for second GEMM ----
    v16h a2;
    #pragma unroll
    for (int hh = 0; hh < 16; ++hh) {
        int k = (hh < 8) ? (kbaseA + hh) : (kbaseA + 8 + hh);
        a2[hh] = (f16)tile[waveId][r][k];
    }

    v16h d0 = wbV[lane];
    v16h d1 = wbV[32 + lane];

    v8f e0 = {}, e1 = {};
    e0 = __builtin_amdgcn_wmma_f32_16x16x32_f16(false, a2, false, d0, (short)0, e0, false, false);
    e1 = __builtin_amdgcn_wmma_f32_16x16x32_f16(false, a2, false, d1, (short)0, e1, false, false);

    // ---- fused bias + BN (scale/shift precomputed), restage, vector store ----
    const float s0 = scale[r], s1 = scale[r + 16];
    const float t0 = shift[r], t1 = shift[r + 16];
    #pragma unroll
    for (int i = 0; i < 8; ++i) {
        tile[waveId][rofsD + i][r]      = e0[i] * s0 + t0;
        tile[waveId][rofsD + i][r + 16] = e1[i] * s1 + t1;
    }

    // each lane stores 16 contiguous floats (half a row) as 4x b128
    const int orow = lane >> 1;
    const int ocol = (lane & 1) * 16;
    const int rowo = row_base + orow;
    if (rowo < N_NODES) {
        v4f*       dst = (v4f*)(hout + (long)rowo * HID + ocol);
        const v4f* src = (const v4f*)(&tile[waveId][orow][ocol]);  // 16B aligned
        #pragma unroll
        for (int q = 0; q < 4; ++q) dst[q] = src[q];
    }
}

// ---------------------------------------------------------------------------
// per-graph mean pool: scatter phase (wave per node, lane = channel)
// ---------------------------------------------------------------------------
__global__ __launch_bounds__(256) void pool_scatter_kernel(const int* __restrict__ batch,
                                                           const float* __restrict__ h,
                                                           float* __restrict__ pooled,
                                                           float* __restrict__ cnt) {
    int wave = (blockIdx.x * blockDim.x + threadIdx.x) >> 5;
    int lane = threadIdx.x & 31;
    if (wave >= N_NODES) return;
    int g = batch[wave];
    atomicAdd(&pooled[(long)g * HID + lane], h[(long)wave * HID + lane]);
    if (lane == 0) atomicAdd(&cnt[g], 1.0f);
}

// ---------------------------------------------------------------------------
// head: mean -> (32x16)+ReLU -> (16x1) -> sigmoid   (256 graphs, tiny)
// ---------------------------------------------------------------------------
__global__ __launch_bounds__(256) void head_kernel(const float* __restrict__ pooled,
                                                   const float* __restrict__ cnt,
                                                   const float* __restrict__ lbw,
                                                   const float* __restrict__ lbb,
                                                   const float* __restrict__ lmw,
                                                   const float* __restrict__ lmb,
                                                   float* __restrict__ out) {
    int g = blockIdx.x * blockDim.x + threadIdx.x;
    if (g >= N_GRAPHS) return;
    float c = fmaxf(cnt[g], 1.0f);
    float p[HID];
    #pragma unroll
    for (int i = 0; i < HID; ++i) p[i] = pooled[(long)g * HID + i] / c;
    float z = lmb[0];
    #pragma unroll
    for (int j = 0; j < DEC; ++j) {
        float s = lbb[j];
        #pragma unroll
        for (int i = 0; i < HID; ++i) s += p[i] * lbw[i * DEC + j];
        z += fmaxf(s, 0.0f) * lmw[j];
    }
    out[g] = 1.0f / (1.0f + expf(-z));
}

// ---------------------------------------------------------------------------
// launcher
// ---------------------------------------------------------------------------
extern "C" void kernel_launch(void* const* d_in, const int* in_sizes, int n_in,
                              void* d_out, int out_size, void* d_ws, size_t ws_size,
                              hipStream_t stream) {
    const float* x     = (const float*)d_in[0];
    const int*   ei    = (const int*)d_in[1];
    const int*   batch = (const int*)d_in[2];

    PrepArgs pa;
    const float* ba[3];
    for (int l = 0; l < 3; ++l) {
        int base = 3 + 8 * l;
        pa.wa[l] = (const float*)d_in[base + 0];
        ba[l]    = (const float*)d_in[base + 1];
        pa.wb[l] = (const float*)d_in[base + 2];
        pa.bb[l] = (const float*)d_in[base + 3];
        pa.g[l]  = (const float*)d_in[base + 4];
        pa.bt[l] = (const float*)d_in[base + 5];
        pa.m[l]  = (const float*)d_in[base + 6];
        pa.v[l]  = (const float*)d_in[base + 7];
        pa.din[l] = (l == 0) ? IN_DIM : HID;
    }
    const float* lbw = (const float*)d_in[27];
    const float* lbb = (const float*)d_in[28];
    const float* lmw = (const float*)d_in[29];
    const float* lmb = (const float*)d_in[30];
    float* out = (float*)d_out;

    // ---- workspace carve-up (d_ws is 256B aligned; FEAT*4 is 256B multiple) ----
    const long FEAT = (long)N_NODES * HID;     // 3.2M floats
    float* hA     = (float*)d_ws;
    float* hB     = hA + FEAT;
    float* agg    = hB + FEAT;
    f16*   waS    = (f16*)(agg + FEAT);
    f16*   wbS    = waS + 3 * HID * HID;
    float* scale  = (float*)(wbS + 3 * HID * HID);
    float* shift  = scale + 3 * HID;
    float* pooled = shift + 3 * HID;           // 256*32
    float* cnt    = pooled + (long)N_GRAPHS * HID;

    // ---- prep: weights -> swizzled f16 fragments, BN fold ----
    prep_kernel<<<3, 256, 0, stream>>>(pa, waS, wbS, scale, shift);

    // ---- pad x into hA ----
    {
        long t = (long)N_NODES * 32;
        pad_x_kernel<<<(int)((t + 255) / 256), 256, 0, stream>>>(x, hA);
    }

    float* hcur = hA;
    float* hnxt = hB;
    const int mlp_blocks = (N_NODES + 127) / 128;   // 8 waves x 16 nodes per block

    for (int l = 0; l < 3; ++l) {
        // agg := h  (so scatter accumulates h + sum of neighbors in one buffer)
        copy_kernel<<<(int)((FEAT + 255) / 256), 256, 0, stream>>>(hcur, agg, FEAT);
        {
            long t = (long)N_EDGES * 32;
            edge_scatter_kernel<<<(int)((t + 255) / 256), 256, 0, stream>>>(ei, hcur, agg);
        }
        gin_mlp_kernel<<<mlp_blocks, 256, 0, stream>>>(agg,
                                                       waS + l * HID * HID,
                                                       wbS + l * HID * HID,
                                                       ba[l],
                                                       scale + l * HID,
                                                       shift + l * HID,
                                                       hnxt);
        float* tmp = hcur; hcur = hnxt; hnxt = tmp;
    }

    // ---- mean pool + head ----
    {
        long nz = (long)N_GRAPHS * HID + N_GRAPHS;   // pooled + cnt contiguous
        zero_kernel<<<(int)((nz + 255) / 256), 256, 0, stream>>>(pooled, nz);
        long t = (long)N_NODES * 32;
        pool_scatter_kernel<<<(int)((t + 255) / 256), 256, 0, stream>>>(batch, hcur, pooled, cnt);
        head_kernel<<<1, 256, 0, stream>>>(pooled, cnt, lbw, lbb, lmw, lmb, out);
    }
    (void)in_sizes; (void)n_in; (void)out_size; (void)ws_size;
}
